// RoboticsSAE_61821759259157
// MI455X (gfx1250) — compile-verified, matrix operand order checked by reading, not run on hardware
//
#include <hip/hip_runtime.h>
#include <hip/hip_bf16.h>

// ---------------------------------------------------------------------------
// TopK SAE forward for MI455X (gfx1250).
//   encode : bf16 WMMA GEMM, 16 rows/block, full 4096-latent tile in LDS
//   top-k  : per-wave iterative selection with wave32 shuffle reduction
//   decode : sparse (16 nnz/row) gather-accumulate from transposed dec_w
// ---------------------------------------------------------------------------

#define D_MODEL   1024
#define D_LATENTS 4096
#define TOPK      16
#define ROWS      16           // rows per block (one WMMA M tile)
#define NTOKENS   (4 * 4096)   // 16384
#define XSTRIDE   1032         // bf16 elements; 2064B row stride (16B aligned, conflict-free)
#define PSTRIDE   4100         // f32 elements; 16400B row stride (16B aligned, +4-bank skew)

typedef __attribute__((ext_vector_type(16))) __bf16          v16bf;
typedef __attribute__((ext_vector_type(8)))  float           v8f;
typedef __attribute__((ext_vector_type(8)))  unsigned short  v8u;
typedef __attribute__((ext_vector_type(16))) unsigned short  v16u;

__device__ __forceinline__ unsigned short f32_to_bf16_rne(float f) {
    unsigned u = __float_as_uint(f);
    unsigned r = u + 0x7FFFu + ((u >> 16) & 1u);   // round-to-nearest-even
    return (unsigned short)(r >> 16);
}

// Build a v16bf WMMA fragment from two 16-byte chunks at p and p+16 elements.
// Per CDNA5 16-bit operand layout: a lane holds K {kb..kb+7} and {kb+16..kb+23}.
__device__ __forceinline__ v16bf load_frag(const unsigned short* p) {
    v8u lo = *(const v8u*)(p);
    v8u hi = *(const v8u*)(p + 16);
    v16u t = __builtin_shufflevector(lo, hi, 0,1,2,3,4,5,6,7,8,9,10,11,12,13,14,15);
    return __builtin_bit_cast(v16bf, t);
}

// ---------------------------------------------------------------------------
// Prep kernel: enc_w f32 -> bf16 (row-major, matches WMMA B-fragment loads)
// ---------------------------------------------------------------------------
__global__ void sae_convert_w_bf16(const float* __restrict__ w,
                                   unsigned short* __restrict__ wbf) {
    int i = blockIdx.x * 256 + threadIdx.x;   // D_LATENTS*D_MODEL / 256 blocks
    wbf[i] = f32_to_bf16_rne(w[i]);
}

// ---------------------------------------------------------------------------
// Prep kernel: dec_w [1024][4096] -> dec_wT [4096][1024] (tiled transpose)
// ---------------------------------------------------------------------------
__global__ void sae_transpose_dec(const float* __restrict__ in,
                                  float* __restrict__ out) {
    __shared__ float tile[32][33];
    int tx = threadIdx.x & 31;
    int ty = threadIdx.x >> 5;                 // 0..7
    int c0 = blockIdx.x * 32;                  // along 4096
    int r0 = blockIdx.y * 32;                  // along 1024
#pragma unroll
    for (int j = 0; j < 32; j += 8)
        tile[ty + j][tx] = in[(size_t)(r0 + ty + j) * D_LATENTS + c0 + tx];
    __syncthreads();
#pragma unroll
    for (int j = 0; j < 32; j += 8)
        out[(size_t)(c0 + ty + j) * D_MODEL + r0 + tx] = tile[tx][ty + j];
}

// ---------------------------------------------------------------------------
// Main fused kernel: encode GEMM (WMMA bf16) + bias + top-16 + scatter + decode
// ---------------------------------------------------------------------------
__global__ __launch_bounds__(256)
void sae_fused_kernel(const float* __restrict__ x,
                      const float* __restrict__ enc_b,
                      const float* __restrict__ enc_b2,
                      const unsigned short* __restrict__ wbf,   // [4096][1024] bf16
                      const float* __restrict__ decT,           // [4096][1024] f32
                      float* __restrict__ recon,                // [16384][1024]
                      float* __restrict__ latents) {            // [16384][4096]
    __shared__ unsigned short xlds[ROWS * XSTRIDE];   //  33.0 KB
    __shared__ float          pre [ROWS * PSTRIDE];   // 262.4 KB
    __shared__ int            topIdx[ROWS][TOPK];     //   1 KB
    __shared__ float          topVal[ROWS][TOPK];     //   1 KB

    const int tid  = threadIdx.x;
    const int lane = tid & 31;
    const int wave = tid >> 5;                // 0..7
    const size_t rowBase = (size_t)blockIdx.x * ROWS;

    // ---- load + convert 16x1024 x-tile into LDS as bf16 -------------------
    const float* xblk = x + rowBase * D_MODEL;
    for (int e = tid; e < ROWS * D_MODEL; e += 256) {
        int r = e >> 10, c = e & (D_MODEL - 1);
        xlds[r * XSTRIDE + c] = f32_to_bf16_rne(xblk[r * D_MODEL + c]);
    }
    __syncthreads();

    // ---- encode GEMM: wave covers 32 N-tiles (512 latents), K = 1024 ------
    const int hi = lane >> 4;          // half-wave selector
    const int mn = lane & 15;          // A: row m   B: col n (within tile)
    const int kb = hi * 8;             // per-lane K base inside a 32-K step
    const unsigned short* xl = &xlds[mn * XSTRIDE + kb];

    for (int tp = 0; tp < 16; ++tp) {
        const int nt0 = wave * 32 + tp * 2;    // N-tile pair shares A fragment
        const int nt1 = nt0 + 1;
        v8f c0 = {}, c1 = {};
        const unsigned short* b0p = wbf + (size_t)(nt0 * 16 + mn) * D_MODEL + kb;
        const unsigned short* b1p = b0p + 16 * D_MODEL;
#pragma unroll 4
        for (int kk = 0; kk < D_MODEL; kk += 32) {
            v16bf a  = load_frag(xl  + kk);
            v16bf b0 = load_frag(b0p + kk);
            v16bf b1 = load_frag(b1p + kk);
            c0 = __builtin_amdgcn_wmma_f32_16x16x32_bf16(false, a, false, b0,
                                                         (short)0, c0, false, false);
            c1 = __builtin_amdgcn_wmma_f32_16x16x32_bf16(false, a, false, b1,
                                                         (short)0, c1, false, false);
        }
        // store C tiles (+ bias) into the LDS pre tile
        {
            int n = nt0 * 16 + mn;
            float bs = enc_b[n] + enc_b2[n];
#pragma unroll
            for (int j = 0; j < 8; ++j)
                pre[(j + hi * 8) * PSTRIDE + n] = c0[j] + bs;
        }
        {
            int n = nt1 * 16 + mn;
            float bs = enc_b[n] + enc_b2[n];
#pragma unroll
            for (int j = 0; j < 8; ++j)
                pre[(j + hi * 8) * PSTRIDE + n] = c1[j] + bs;
        }
    }
    __syncthreads();

    // ---- per-row top-16 (wave owns rows 2w, 2w+1) --------------------------
    const float NEG = -__builtin_inff();
    for (int rr = 0; rr < 2; ++rr) {
        const int r = wave * 2 + rr;
        float* prow = &pre[r * PSTRIDE];
        for (int s = 0; s < TOPK; ++s) {
            float best = NEG;
            int   bidx = 0x7FFFFFFF;
            for (int i = lane; i < D_LATENTS; i += 32) {
                float v = prow[i];
                if (v > best) { best = v; bidx = i; }
            }
#pragma unroll
            for (int off = 16; off > 0; off >>= 1) {
                float ov = __shfl_xor(best, off, 32);
                int   oi = __shfl_xor(bidx, off, 32);
                if (ov > best || (ov == best && oi < bidx)) { best = ov; bidx = oi; }
            }
            if (lane == 0) {
                topIdx[r][s] = bidx;
                topVal[r][s] = fmaxf(best, 0.0f);   // ReLU on selected values
            }
            if (lane == (bidx & 31)) prow[bidx] = NEG;  // claim winner
        }
        // rebuild row as the sparse latents tile: zeros + scattered ReLU vals
        for (int i = lane; i < D_LATENTS; i += 32) prow[i] = 0.0f;
        if (lane < TOPK) prow[topIdx[r][lane]] = topVal[r][lane];
    }

    // ---- sparse decode: recon[r] = sum_s val_s * decT[idx_s][:] ------------
    for (int rr = 0; rr < 2; ++rr) {
        const int r = wave * 2 + rr;
        float4 acc[8];
#pragma unroll
        for (int j = 0; j < 8; ++j) acc[j] = make_float4(0.f, 0.f, 0.f, 0.f);
#pragma unroll
        for (int s = 0; s < TOPK; ++s) {
            const int   ci = topIdx[r][s];
            const float cv = topVal[r][s];
            const float4* col = (const float4*)(decT + (size_t)ci * D_MODEL);
#pragma unroll
            for (int j = 0; j < 8; ++j) {
                float4 w4 = col[lane + 32 * j];
                acc[j].x += cv * w4.x;  acc[j].y += cv * w4.y;
                acc[j].z += cv * w4.z;  acc[j].w += cv * w4.w;
            }
        }
        float4* outp = (float4*)(recon + (rowBase + r) * D_MODEL);
#pragma unroll
        for (int j = 0; j < 8; ++j) outp[lane + 32 * j] = acc[j];
    }

    // ---- stream latents tile to HBM (coalesced float4) ---------------------
    __syncthreads();
    for (int f = tid; f < ROWS * D_LATENTS / 4; f += 256) {
        int r  = f >> 10;          // 1024 float4 per row
        int c4 = f & 1023;
        float4 v = *(const float4*)&pre[r * PSTRIDE + c4 * 4];
        *(float4*)(latents + (rowBase + r) * (size_t)D_LATENTS + c4 * 4) = v;
    }
}

// ---------------------------------------------------------------------------
extern "C" void kernel_launch(void* const* d_in, const int* in_sizes, int n_in,
                              void* d_out, int out_size, void* d_ws, size_t ws_size,
                              hipStream_t stream) {
    const float* x       = (const float*)d_in[0];   // [4,4096,1024]
    const float* enc_w   = (const float*)d_in[1];   // [4096,1024]
    const float* enc_b   = (const float*)d_in[2];   // [4096]
    const float* enc_b2  = (const float*)d_in[3];   // [4096] (encoder_bias)
    const float* dec_w   = (const float*)d_in[4];   // [1024,4096]

    float* recon   = (float*)d_out;                                  // [16384,1024]
    float* latents = recon + (size_t)NTOKENS * D_MODEL;              // [16384,4096]

    unsigned short* wbf = (unsigned short*)d_ws;                                  // 8 MB
    float* decT = (float*)((char*)d_ws + (size_t)D_LATENTS * D_MODEL * 2);        // 16 MB

    // enc_w -> bf16
    sae_convert_w_bf16<<<(D_LATENTS * D_MODEL) / 256, 256, 0, stream>>>(enc_w, wbf);
    // dec_w -> dec_wT
    sae_transpose_dec<<<dim3(D_LATENTS / 32, D_MODEL / 32), 256, 0, stream>>>(dec_w, decT);
    // fused encode + topk + scatter + decode
    sae_fused_kernel<<<NTOKENS / ROWS, 256, 0, stream>>>(x, enc_b, enc_b2, wbf, decT,
                                                         recon, latents);
}
